// LstmBlock_5351529251116
// MI455X (gfx1250) — matrix operand order repaired and checked by reading, not compiled, not run on hardware
//
#include <hip/hip_runtime.h>

// ---------------------------------------------------------------------------
// LSTM block for MI455X (gfx1250): bf16 WMMA GEMMs + persistent recurrence.
//   B=32, S=512, D=1024, H=1024, 4H=4096
//   - batch GEMM xU = x@U+b via v_wmma_f32_16x16x32_bf16
//   - recurrence: V cached in LDS via TDM (tensor_load_to_lds), h@V via WMMA,
//     cell state c held in accumulator-layout registers for all 512 steps.
// ---------------------------------------------------------------------------

typedef __attribute__((ext_vector_type(16))) __bf16 v16bf;
typedef __attribute__((ext_vector_type(8)))  __bf16 v8bf;
typedef __attribute__((ext_vector_type(8)))  float  v8f;
typedef __attribute__((ext_vector_type(4)))  unsigned u32x4;
typedef __attribute__((ext_vector_type(8)))  int      i32x8;
typedef __attribute__((ext_vector_type(4)))  int      i32x4;

union ABReg { v16bf v; v8bf h[2]; };

constexpr int BATCH = 32;
constexpr int SEQ   = 512;
constexpr int DIM   = 1024;
constexpr int HID   = 1024;
constexpr int G4    = 4096;          // 4*HID
constexpr int NWG_REC = 32;          // 64 column-tiles / 2 waves per WG
constexpr int REC_THREADS = 64;      // 2 wave32s

#if defined(__has_builtin)
#  if __has_builtin(__builtin_amdgcn_tensor_load_to_lds)
#    define USE_TDM 1
#  endif
#  if __has_builtin(__builtin_amdgcn_s_wait_tensorcnt)
#    define HAVE_WAIT_TENSOR 1
#  endif
#endif
#ifndef USE_TDM
#  define USE_TDM 0
#endif
#ifndef HAVE_WAIT_TENSOR
#  define HAVE_WAIT_TENSOR 0
#endif

__device__ __forceinline__ __bf16 f2bf(float f) {
    unsigned u = __builtin_bit_cast(unsigned, f);
    u = (u + 0x7FFFu + ((u >> 16) & 1u)) >> 16;      // round-to-nearest-even
    unsigned short s = (unsigned short)u;
    return __builtin_bit_cast(__bf16, s);
}

__device__ __forceinline__ float fsig(float x) {
    x = fminf(fmaxf(x, -30.f), 30.f);
    return 1.0f / (1.0f + __expf(-x));
}
__device__ __forceinline__ float ftanh(float x) {
    x = fminf(fmaxf(x, -15.f), 15.f);
    float e = __expf(2.0f * x);
    return (e - 1.0f) / (e + 1.0f);
}

#if USE_TDM
// Flat 1-D tile copy global->LDS through the Tensor Data Mover.
// D# per CDNA5 ISA ch.8: group0 = {count, lds_addr, global_addr, type=2},
// group1 = {data_size=2B, tensor_dim0=tile_dim0=nelem, tensor_dim1=tile_dim1=1}.
__device__ __forceinline__ void tdm_copy_b16(const void* gsrc, unsigned lds_byte_addr,
                                             unsigned nelem) {
    unsigned long long ga = (unsigned long long)(uintptr_t)gsrc;
    u32x4 g0;
    g0[0] = 1u;                                           // count=1 (valid user D#)
    g0[1] = lds_byte_addr;                                // lds_addr [63:32]
    g0[2] = (unsigned)(ga & 0xffffffffu);                 // global_addr lo
    g0[3] = (unsigned)((ga >> 32) & 0x01ffffffu)          // global_addr hi (57b)
            | (2u << 30);                                 // type=2 ("image")
    i32x8 g1;
    g1[0] = (int)(1u << 16);                              // data_size=1 -> 2 bytes
    g1[1] = (int)((nelem & 0xffffu) << 16);               // tensor_dim0[15:0]  @ bits 63:48
    g1[2] = (int)(((nelem >> 16) & 0xffffu)               // tensor_dim0[31:16] @ bits 79:64
                  | (1u << 16));                          // tensor_dim1=1      @ bits 95:80
    g1[3] = (int)((nelem & 0xffffu) << 16);               // tile_dim0          @ bits 127:112
    g1[4] = 1;                                            // tile_dim1=1 (tile_dim2=0)
    g1[5] = (int)nelem;                                   // tensor_dim0_stride lo
    g1[6] = 0;
    g1[7] = 0;
    i32x4 z4 = {0, 0, 0, 0};
#if __clang_major__ >= 23
    i32x8 z8 = {0, 0, 0, 0, 0, 0, 0, 0};
    __builtin_amdgcn_tensor_load_to_lds(g0, g1, z4, z4, z8, 0);
#else
    __builtin_amdgcn_tensor_load_to_lds(g0, g1, z4, z4, 0);
#endif
}
#endif

// ------------------------------- pack kernels ------------------------------

__global__ void k_pack_x(const float* __restrict__ x, __bf16* __restrict__ xb, int n) {
    int i = blockIdx.x * blockDim.x + threadIdx.x;
    if (i < n) xb[i] = f2bf(x[i]);
}

// dst[n][k] (K-contiguous) <- concat_g srcg[k][j],  n = g*HID + j
__global__ void k_pack_gateT(const float* __restrict__ s0, const float* __restrict__ s1,
                             const float* __restrict__ s2, const float* __restrict__ s3,
                             __bf16* __restrict__ dst, int krows) {
    int i = blockIdx.x * blockDim.x + threadIdx.x;   // over 4096*krows
    if (i >= G4 * krows) return;
    int n = i / krows, k = i % krows;
    const float* src = (n < HID) ? s0 : (n < 2 * HID) ? s1 : (n < 3 * HID) ? s2 : s3;
    int j = n & (HID - 1);
    dst[i] = f2bf(src[(size_t)k * HID + j]);
}

__global__ void k_pack_bias(const float* __restrict__ b0, const float* __restrict__ b1,
                            const float* __restrict__ b2, const float* __restrict__ b3,
                            float* __restrict__ dst) {
    int i = blockIdx.x * blockDim.x + threadIdx.x;
    if (i < G4) {
        const float* s = (i < HID) ? b0 : (i < 2 * HID) ? b1 : (i < 3 * HID) ? b2 : b3;
        dst[i] = s[i & (HID - 1)];
    }
}

__global__ void k_pack_h0(const float* __restrict__ h0, __bf16* __restrict__ hb, int n) {
    int i = blockIdx.x * blockDim.x + threadIdx.x;
    if (i < n) hb[i] = f2bf(h0[i]);
}

__global__ void k_init_sync(unsigned* __restrict__ s) {
    if (threadIdx.x == 0 && blockIdx.x == 0) { s[0] = 0u; s[1] = 0u; }
}

// ---------------------- batch GEMM: xU = x @ U + b -------------------------
// xb: [B*S=16384][1024] bf16 row-major,  ubT: [4096][1024] bf16 (K-contig),
// xu: [16384][4096] fp32.  Wave computes a 32x16 C tile (2 WMMA M-tiles).
__global__ void k_gemm_xu(const __bf16* __restrict__ xb,
                          const __bf16* __restrict__ ubT,
                          const float*  __restrict__ bias,
                          float* __restrict__ xu) {
    const int lane = threadIdx.x & 31;
    const int wi   = threadIdx.x >> 5;
    const int mb   = blockIdx.x;               // 0..511 -> rows mb*32..+32
    const int nt   = blockIdx.y * 8 + wi;      // 0..255 -> cols nt*16..+16
    const int nl   = lane & 15;
    const int hi   = lane >> 4;                // lane half (A/B layout split)

    v8f acc0 = {0, 0, 0, 0, 0, 0, 0, 0};
    v8f acc1 = acc0;

    const __bf16* arow0 = xb + (size_t)(mb * 32 + nl) * DIM;
    const __bf16* arow1 = arow0 + (size_t)16 * DIM;
    const __bf16* bcol  = ubT + (size_t)(nt * 16 + nl) * DIM;

    for (int kt = 0; kt < DIM / 32; ++kt) {
        const int kbA = kt * 32 + hi * 8;      // A: lanes 0-15 K 0..7/16..23; 16-31 +8
        const int kbB = kt * 32 + hi * 8;      // B: lanes 0-15 K 0..7/16..23; 16-31 +8
        ABReg a0, a1, b;
        a0.h[0] = *(const v8bf*)(arow0 + kbA);
        a0.h[1] = *(const v8bf*)(arow0 + kbA + 16);
        a1.h[0] = *(const v8bf*)(arow1 + kbA);
        a1.h[1] = *(const v8bf*)(arow1 + kbA + 16);
        b.h[0]  = *(const v8bf*)(bcol + kbB);
        b.h[1]  = *(const v8bf*)(bcol + kbB + 16);
        acc0 = __builtin_amdgcn_wmma_f32_16x16x32_bf16(false, a0.v, false, b.v,
                                                       (short)0, acc0, false, false);
        acc1 = __builtin_amdgcn_wmma_f32_16x16x32_bf16(false, a1.v, false, b.v,
                                                       (short)0, acc1, false, false);
    }

    const float bv = bias[nt * 16 + nl];
    #pragma unroll
    for (int e = 0; e < 8; ++e) {
        const int m = hi * 8 + e;              // C layout: elem e, lane half -> row
        xu[(size_t)(mb * 32 + m)      * G4 + nt * 16 + nl] = acc0[e] + bv;
        xu[(size_t)(mb * 32 + 16 + m) * G4 + nt * 16 + nl] = acc1[e] + bv;
    }
}

// --------------------------- grid-wide barrier -----------------------------
__device__ __forceinline__ void grid_barrier(unsigned* sync, unsigned nwg) {
    __threadfence();
    __syncthreads();
    if (threadIdx.x == 0) {
        unsigned g = __hip_atomic_load(&sync[1], __ATOMIC_RELAXED, __HIP_MEMORY_SCOPE_AGENT);
        unsigned a = __hip_atomic_fetch_add(&sync[0], 1u, __ATOMIC_ACQ_REL, __HIP_MEMORY_SCOPE_AGENT);
        if (a == nwg - 1u) {
            __hip_atomic_store(&sync[0], 0u, __ATOMIC_RELAXED, __HIP_MEMORY_SCOPE_AGENT);
            __hip_atomic_fetch_add(&sync[1], 1u, __ATOMIC_RELEASE, __HIP_MEMORY_SCOPE_AGENT);
        } else {
            while (__hip_atomic_load(&sync[1], __ATOMIC_ACQUIRE, __HIP_MEMORY_SCOPE_AGENT) == g)
                __builtin_amdgcn_s_sleep(1);
        }
    }
    __syncthreads();
}

// ----------------------- persistent recurrent kernel -----------------------
// 32 WGs x 2 waves; wave owns H-columns [j0, j0+16) across all 4 gates.
// V stripe (4 gates x 16 cols x 1024 K, bf16, K-contig) cached in LDS (128KB/wave),
// staged by the Tensor Data Mover. c-state kept in registers across all 512 steps.
__global__ void k_lstm_rec(const float* __restrict__ xu,    // [B*S][4096]
                           const __bf16* __restrict__ vbT,  // [4096][1024]
                           const float* __restrict__ c0,    // [32][1024]
                           __bf16* __restrict__ hbuf,       // [2][32][1024]
                           float* __restrict__ hs,          // [32][512][1024]
                           float* __restrict__ ht,          // [32][1024]
                           float* __restrict__ ct,          // [32][1024]
                           unsigned* __restrict__ sync) {
    extern __shared__ char smem_raw[];
    __bf16* ldsV = (__bf16*)smem_raw;          // [2 waves][64 cols][1024]

    const int lane = threadIdx.x & 31;
    const int wi   = threadIdx.x >> 5;         // 0..1
    const int wid  = blockIdx.x * 2 + wi;      // 0..63
    const int j0   = wid * 16;
    const int nl   = lane & 15;
    const int hi   = lane >> 4;

#if USE_TDM
    // ---- one-time V stripe preload via Tensor Data Mover ----
    // Each wave DMAs its 4 contiguous 32KB stripes (16 cols x 1024 K, bf16).
    {
        unsigned ldsBase = (unsigned)(uintptr_t)ldsV;   // LDS aperture: offset in addr[31:0]
        #pragma unroll
        for (int g = 0; g < 4; ++g) {
            const int cc0 = wi * 64 + g * 16;                       // block-local col
            const __bf16* src = vbT + (size_t)(g * HID + wid * 16) * HID;
            tdm_copy_b16(src, ldsBase + (unsigned)cc0 * HID * 2u, 16 * HID);
        }
#if HAVE_WAIT_TENSOR
        __builtin_amdgcn_s_wait_tensorcnt(0);
#endif
    }
#else
    // ---- fallback: manual coalesced b128 copies ----
    for (int fc = threadIdx.x; fc < 128 * 128; fc += blockDim.x) {
        int cc = fc >> 7;                      // block-local col 0..127
        int ch = fc & 127;                     // 8-elem chunk within col
        int w  = cc >> 6;
        int g  = (cc >> 4) & 3;
        int nn = cc & 15;
        int gcol = g * HID + (blockIdx.x * 2 + w) * 16 + nn;
        *(v8bf*)(ldsV + (size_t)cc * HID + ch * 8) =
            *(const v8bf*)(vbT + (size_t)gcol * HID + ch * 8);
    }
#endif
    __syncthreads();

    // ---- cell state in registers (C-matrix layout) ----
    float creg[2][8], hreg[2][8];
    #pragma unroll
    for (int mt = 0; mt < 2; ++mt)
        #pragma unroll
        for (int e = 0; e < 8; ++e) {
            int m = mt * 16 + hi * 8 + e;
            creg[mt][e] = c0[(size_t)m * HID + j0 + nl];
            hreg[mt][e] = 0.f;
        }

    const __bf16* ldsW = ldsV + (size_t)wi * 64 * HID;

    for (int s = 0; s < SEQ; ++s) {
        const __bf16* hcur = hbuf + (size_t)(s & 1) * BATCH * HID;
        __bf16* hnxt       = hbuf + (size_t)((s + 1) & 1) * BATCH * HID;

        v8f acc[4][2];
        #pragma unroll
        for (int g = 0; g < 4; ++g) {
            acc[g][0] = (v8f){0, 0, 0, 0, 0, 0, 0, 0};
            acc[g][1] = acc[g][0];
        }

        const __bf16* ar0 = hcur + (size_t)nl * HID;
        const __bf16* ar1 = ar0 + (size_t)16 * HID;

        for (int kt = 0; kt < HID / 32; ++kt) {
            const int kbA = kt * 32 + hi * 8;
            const int kbB = kt * 32 + hi * 8;
            ABReg a0, a1;
            a0.h[0] = *(const v8bf*)(ar0 + kbA);
            a0.h[1] = *(const v8bf*)(ar0 + kbA + 16);
            a1.h[0] = *(const v8bf*)(ar1 + kbA);
            a1.h[1] = *(const v8bf*)(ar1 + kbA + 16);
            #pragma unroll
            for (int g = 0; g < 4; ++g) {
                ABReg bm;
                const __bf16* bp = ldsW + (size_t)(g * 16 + nl) * HID + kbB;
                bm.h[0] = *(const v8bf*)(bp);
                bm.h[1] = *(const v8bf*)(bp + 16);
                acc[g][0] = __builtin_amdgcn_wmma_f32_16x16x32_bf16(
                    false, a0.v, false, bm.v, (short)0, acc[g][0], false, false);
                acc[g][1] = __builtin_amdgcn_wmma_f32_16x16x32_bf16(
                    false, a1.v, false, bm.v, (short)0, acc[g][1], false, false);
            }
        }

        // ---- gates + cell update (all per-lane, no cross-wave traffic) ----
        #pragma unroll
        for (int mt = 0; mt < 2; ++mt) {
            #pragma unroll
            for (int e = 0; e < 8; ++e) {
                const int m = mt * 16 + hi * 8 + e;     // batch index
                const size_t xrow = ((size_t)m * SEQ + s) * G4;
                const int col = j0 + nl;
                float zi = acc[0][mt][e] + xu[xrow + 0 * HID + col];
                float zf = acc[1][mt][e] + xu[xrow + 1 * HID + col];
                float zg = acc[2][mt][e] + xu[xrow + 2 * HID + col];
                float zo = acc[3][mt][e] + xu[xrow + 3 * HID + col];
                float iv = fsig(zi), fv = fsig(zf), gv = ftanh(zg), ov = fsig(zo);
                float cv = fv * creg[mt][e] + iv * gv;
                float hv = ov * ftanh(cv);
                creg[mt][e] = cv;
                hreg[mt][e] = hv;
                hs[((size_t)m * SEQ + s) * HID + col] = hv;
                hnxt[(size_t)m * HID + col] = f2bf(hv);
            }
        }

        if (s + 1 < SEQ) {  // warm L2 for next step's xU slice
            const size_t xrow = ((size_t)(hi * 8) * SEQ + (s + 1)) * G4;
            __builtin_prefetch(&xu[xrow + j0 + nl], 0, 0);
        }

        grid_barrier(sync, NWG_REC);
    }

    #pragma unroll
    for (int mt = 0; mt < 2; ++mt)
        #pragma unroll
        for (int e = 0; e < 8; ++e) {
            const int m = mt * 16 + hi * 8 + e;
            ht[(size_t)m * HID + j0 + nl] = hreg[mt][e];
            ct[(size_t)m * HID + j0 + nl] = creg[mt][e];
        }
}

// ------------------------------- launcher ----------------------------------

extern "C" void kernel_launch(void* const* d_in, const int* in_sizes, int n_in,
                              void* d_out, int out_size, void* d_ws, size_t ws_size,
                              hipStream_t stream) {
    (void)in_sizes; (void)n_in; (void)out_size; (void)ws_size;

    const float* x  = (const float*)d_in[0];
    const float* h0 = (const float*)d_in[1];
    const float* c0 = (const float*)d_in[2];
    const float* U0 = (const float*)d_in[3],  *V0 = (const float*)d_in[4],  *b0 = (const float*)d_in[5];
    const float* U1 = (const float*)d_in[6],  *V1 = (const float*)d_in[7],  *b1 = (const float*)d_in[8];
    const float* U2 = (const float*)d_in[9],  *V2 = (const float*)d_in[10], *b2 = (const float*)d_in[11];
    const float* U3 = (const float*)d_in[12], *V3 = (const float*)d_in[13], *b3 = (const float*)d_in[14];

    float* out = (float*)d_out;
    float* hs = out;                                   // [32][512][1024]
    float* ht = out + (size_t)BATCH * SEQ * HID;       // [32][1024]
    float* ct = ht + (size_t)BATCH * HID;              // [32][1024]

    char* ws = (char*)d_ws;
    __bf16*   xbf  = (__bf16*)(ws);                          // 32 MB
    __bf16*   ubT  = (__bf16*)(ws + 33554432);               //  8 MB
    __bf16*   vbT  = (__bf16*)(ws + 41943040);               //  8 MB
    float*    bfv  = (float*) (ws + 50331648);               // 16 KB
    float*    xu   = (float*) (ws + 50348032);               // 256 MB
    __bf16*   hbuf = (__bf16*)(ws + 318783488);              // 128 KB (double-buffered h)
    unsigned* sync = (unsigned*)(ws + 318914560);            // barrier {count, generation}

    const int nx = BATCH * SEQ * DIM;                        // 16,777,216
    k_pack_x<<<(nx + 255) / 256, 256, 0, stream>>>(x, xbf, nx);

    const int nw = G4 * DIM;                                 // 4,194,304
    k_pack_gateT<<<(nw + 255) / 256, 256, 0, stream>>>(U0, U1, U2, U3, ubT, DIM);
    k_pack_gateT<<<(nw + 255) / 256, 256, 0, stream>>>(V0, V1, V2, V3, vbT, HID);
    k_pack_bias<<<(G4 + 255) / 256, 256, 0, stream>>>(b0, b1, b2, b3, bfv);

    const int nh = BATCH * HID;                              // 32,768
    k_pack_h0<<<(nh + 255) / 256, 256, 0, stream>>>(h0, hbuf, nh);
    k_init_sync<<<1, 32, 0, stream>>>(sync);

    // xU = x @ U + b : grid (M/32, N/128), 8 waves per block
    k_gemm_xu<<<dim3(BATCH * SEQ / 32, G4 / 128), 256, 0, stream>>>(xbf, ubT, bfv, xu);

    // persistent recurrence: 32 WGs x 64 threads, 256 KB dynamic LDS per WG
    k_lstm_rec<<<NWG_REC, REC_THREADS, 2 * 64 * HID * sizeof(__bf16), stream>>>(
        xu, vbT, c0, hbuf, hs, ht, ct, sync);
}
